// PointConvK_51170240364924
// MI455X (gfx1250) — matrix operand order repaired and checked by reading, not compiled, not run on hardware
//
#include <hip/hip_runtime.h>
#include <hip/hip_bf16.h>

typedef __attribute__((ext_vector_type(16))) _Float16 v16h;
typedef __attribute__((ext_vector_type(8)))  float    v8f;

#define B_    8
#define N_    4096
#define K_    16
#define D_    61
#define CIN_  64
#define OUT_  64
#define LEAKY_ 0.1f
#define EPS_   1e-5f

__device__ __forceinline__ float leakyf(float x) { return x >= 0.f ? x : LEAKY_ * x; }

// ---------------------------------------------------------------------------
// Kernel 1: fused distance + KNN (no N x N matrix ever materialized).
// One block per (batch, 256-point tile); all 4096 xyz of the batch in LDS.
// ---------------------------------------------------------------------------
__global__ __launch_bounds__(256) void knn_kernel(const float* __restrict__ xyz,
                                                  int* __restrict__ knn) {
  __shared__ float xs[N_], ys[N_], zs[N_];
  const int b    = blockIdx.x / (N_ / 256);
  const int tile = blockIdx.x % (N_ / 256);
  const float* xb = xyz + (size_t)b * 3 * N_;
  for (int i = threadIdx.x; i < N_; i += 256) {
    xs[i] = xb[i];
    ys[i] = xb[N_ + i];
    zs[i] = xb[2 * N_ + i];
  }
  __syncthreads();

  const int n = tile * 256 + threadIdx.x;
  const float qx = xs[n], qy = ys[n], qz = zs[n];

  float bd[K_]; int bi[K_];
#pragma unroll
  for (int i = 0; i < K_; ++i) { bd[i] = 3.4e38f; bi[i] = 0; }

  for (int m = 0; m < N_; ++m) {
    const float dx = xs[m] - qx, dy = ys[m] - qy, dz = zs[m] - qz;
    const float d = dx * dx + dy * dy + dz * dz;
    if (d < bd[K_ - 1]) {
      int pos = K_ - 1;
#pragma unroll
      for (int i = K_ - 1; i >= 1; --i) {
        if (bd[i - 1] > d) { bd[i] = bd[i - 1]; bi[i] = bi[i - 1]; pos = i - 1; }
      }
      bd[pos] = d; bi[pos] = m;
    }
  }
  int* out = knn + ((size_t)b * N_ + n) * K_;
#pragma unroll
  for (int i = 0; i < K_; ++i) out[i] = bi[i];
}

// ---------------------------------------------------------------------------
// Kernel 2: per-point gather + ker = np(16x64) x Wk^T via v_wmma_f32_16x16x32_f16
// (4 OUT tiles x 2 K-chunks = 8 WMMA / point), fused BN+leaky, then the
// algebraically-collapsed aggregation s[o] = sum_k ker_act[k,o] * (np[k,:].Wa),
// second BN+leaky -> aggS[point][64].
// One wave per point, 8 waves per block.
// ---------------------------------------------------------------------------
__global__ __launch_bounds__(256) void pointker_kernel(
    const float* __restrict__ xyz, const float* __restrict__ points,
    const float* __restrict__ Wk, const float* __restrict__ bnk_g,
    const float* __restrict__ bnk_b, const float* __restrict__ bnk_m,
    const float* __restrict__ bnk_v, const float* __restrict__ Wa,
    const float* __restrict__ bna_g, const float* __restrict__ bna_b,
    const float* __restrict__ bna_m, const float* __restrict__ bna_v,
    const int* __restrict__ knn, float* __restrict__ aggS) {
  __shared__ float lds_np[8][K_ * 68];   // row stride 68 floats: conflict-free
  __shared__ float lds_t[8][K_];

  const int w    = threadIdx.x >> 5;
  const int lane = threadIdx.x & 31;
  const int hi   = lane >> 4;            // lane half
  const int l15  = lane & 15;
  const int g    = blockIdx.x * 8 + w;   // global point id
  const int b    = g / N_;
  const int n    = g % N_;

  // ---- gather new_points row k = l15; lo lanes: c in [0,32), hi lanes: [32,64)
  const int k = l15;
  const int j = knn[(size_t)g * K_ + k];
  const float* xb = xyz + (size_t)b * 3 * N_;
  const float* pb = points + (size_t)b * D_ * N_;
  float* row = &lds_np[w][k * 68];
  float tpart = 0.f;
#pragma unroll 8
  for (int cc = 0; cc < 32; ++cc) {
    const int c = hi * 32 + cc;
    float v;
    if (c < 3) v = xb[c * N_ + j] - xb[c * N_ + n];
    else       v = pb[(c - 3) * N_ + j];
    row[c] = v;
    tpart += v * Wa[c];
  }
  tpart += __shfl_xor(tpart, 16);        // t[k] = np[k,:].Wa
  if (lane < K_) lds_t[w][lane] = tpart;
  __syncthreads();

  // ---- A fragments (contiguous k order, same as B): M = l15, K = 16*hi + e
  v16h a0, a1;
  const float* npw = lds_np[w];
#pragma unroll
  for (int e = 0; e < 16; ++e) {
    const int kk = hi * 16 + e;
    a0[e] = (_Float16)npw[l15 * 68 + kk];
    a1[e] = (_Float16)npw[l15 * 68 + kk + 32];
  }
  float tv[8];
#pragma unroll
  for (int r = 0; r < 8; ++r) tv[r] = lds_t[w][r + 8 * hi];

  const float bnaM = bna_m[0];
  const float bnaS = bna_g[0] * rsqrtf(bna_v[0] + EPS_);
  const float bnaB = bna_b[0];

#pragma unroll
  for (int tn = 0; tn < 4; ++tn) {
    const int o = tn * 16 + l15;                 // OUT channel = B/D column
    const float* wkrow = Wk + o * CIN_;
    v16h b0, b1;                                 // B layout: N=l15, K=16*hi+e
#pragma unroll
    for (int e = 0; e < 16; ++e) {
      b0[e] = (_Float16)wkrow[hi * 16 + e];
      b1[e] = (_Float16)wkrow[32 + hi * 16 + e];
    }
    v8f acc = {};
    acc = __builtin_amdgcn_wmma_f32_16x16x32_f16(false, a0, false, b0, (short)0, acc, false, false);
    acc = __builtin_amdgcn_wmma_f32_16x16x32_f16(false, a1, false, b1, (short)0, acc, false, false);

    const float sc = bnk_g[o] * rsqrtf(bnk_v[o] + EPS_);
    const float sh = bnk_b[o] - bnk_m[o] * sc;
    float sp = 0.f;
#pragma unroll
    for (int r = 0; r < 8; ++r) {                // D: M = r + 8*hi = neighbor k
      const float kv = leakyf(acc[r] * sc + sh);
      sp += kv * tv[r];
    }
    sp += __shfl_xor(sp, 16);                    // sum both k-halves
    const float a_o = leakyf((sp - bnaM) * bnaS + bnaB);
    if (hi == 0) aggS[(size_t)g * OUT_ + o] = a_o;
  }
}

// ---------------------------------------------------------------------------
// Kernel 3: out[b,o',n] = leaky(aggS[b,n,:] . Wl[o',:] + bl[o'])  via WMMA.
// One wave per 16 consecutive points; D rows are contiguous in n -> float4 stores.
// ---------------------------------------------------------------------------
__global__ __launch_bounds__(256) void final_linear_kernel(
    const float* __restrict__ aggS, const float* __restrict__ Wl,
    const float* __restrict__ bl, float* __restrict__ out) {
  const int w    = threadIdx.x >> 5;
  const int lane = threadIdx.x & 31;
  const int hi   = lane >> 4;
  const int l15  = lane & 15;
  const int p0   = (blockIdx.x * 8 + w) * 16;    // first of 16 points (b-aligned)
  const int b    = p0 / N_;
  const int n0   = p0 % N_;

  v16h a0, a1;
  const float* arow = aggS + (size_t)(p0 + l15) * OUT_;
#pragma unroll
  for (int e = 0; e < 16; ++e) {
    const int kk = hi * 16 + e;
    a0[e] = (_Float16)arow[kk];
    a1[e] = (_Float16)arow[kk + 32];
  }

#pragma unroll
  for (int tn = 0; tn < 4; ++tn) {
    const int op = tn * 16 + l15;
    const float* wlrow = Wl + op * OUT_;
    v16h b0, b1;
#pragma unroll
    for (int e = 0; e < 16; ++e) {
      b0[e] = (_Float16)wlrow[hi * 16 + e];
      b1[e] = (_Float16)wlrow[32 + hi * 16 + e];
    }
    v8f acc = {};
    acc = __builtin_amdgcn_wmma_f32_16x16x32_f16(false, a0, false, b0, (short)0, acc, false, false);
    acc = __builtin_amdgcn_wmma_f32_16x16x32_f16(false, a1, false, b1, (short)0, acc, false, false);

    const float bias = bl[op];
    float* obase = out + (size_t)b * OUT_ * N_ + (size_t)op * N_ + n0 + hi * 8;
    const float4 lo4 = make_float4(leakyf(acc[0] + bias), leakyf(acc[1] + bias),
                                   leakyf(acc[2] + bias), leakyf(acc[3] + bias));
    const float4 hi4 = make_float4(leakyf(acc[4] + bias), leakyf(acc[5] + bias),
                                   leakyf(acc[6] + bias), leakyf(acc[7] + bias));
    *(float4*)(obase)     = lo4;
    *(float4*)(obase + 4) = hi4;
  }
}

// ---------------------------------------------------------------------------
extern "C" void kernel_launch(void* const* d_in, const int* in_sizes, int n_in,
                              void* d_out, int out_size, void* d_ws, size_t ws_size,
                              hipStream_t stream) {
  const float* xyz    = (const float*)d_in[0];
  const float* points = (const float*)d_in[1];
  const float* Wk     = (const float*)d_in[2];
  const float* bnk_g  = (const float*)d_in[3];
  const float* bnk_b  = (const float*)d_in[4];
  const float* bnk_m  = (const float*)d_in[5];
  const float* bnk_v  = (const float*)d_in[6];
  const float* Wa     = (const float*)d_in[7];
  const float* bna_g  = (const float*)d_in[8];
  const float* bna_b  = (const float*)d_in[9];
  const float* bna_m  = (const float*)d_in[10];
  const float* bna_v  = (const float*)d_in[11];
  const float* Wl     = (const float*)d_in[12];
  const float* bl     = (const float*)d_in[13];
  float* out = (float*)d_out;

  // workspace: [knn idx: B*N*K ints][aggS: B*N*64 floats]  (~10 MB total)
  int*   knn  = (int*)d_ws;
  float* aggS = (float*)((char*)d_ws + (size_t)B_ * N_ * K_ * sizeof(int));

  knn_kernel<<<B_ * (N_ / 256), 256, 0, stream>>>(xyz, knn);
  pointker_kernel<<<(B_ * N_) / 8, 256, 0, stream>>>(
      xyz, points, Wk, bnk_g, bnk_b, bnk_m, bnk_v,
      Wa, bna_g, bna_b, bna_m, bna_v, knn, aggS);
  final_linear_kernel<<<(B_ * N_) / 128, 256, 0, stream>>>(aggS, Wl, bl, out);
}
